// CausalSelfAttention_13185549599239
// MI455X (gfx1250) — matrix operand order repaired and checked by reading, not compiled, not run on hardware
//
#include <hip/hip_runtime.h>
#include <hip/hip_bf16.h>

// ---------------------------------------------------------------------------
// Types matching the gfx1250 WMMA builtin signatures
// ---------------------------------------------------------------------------
typedef __bf16 bf16_t;
typedef __attribute__((ext_vector_type(16))) __bf16 v16bf;
typedef __attribute__((ext_vector_type(8)))  __bf16 v8bf;
typedef __attribute__((ext_vector_type(4)))  __bf16 v4bf;
typedef __attribute__((ext_vector_type(8)))  float  v8f;

__device__ __forceinline__ v8f wmma_bf16(v16bf a, v16bf b, v8f c) {
  // D = A(16x32) * B(32x16) + C(16x16 f32)
  return __builtin_amdgcn_wmma_f32_16x16x32_bf16(
      /*neg_a=*/false, a, /*neg_b=*/false, b,
      /*c_mod=*/(short)0, c, /*reuse_a=*/false, /*reuse_b=*/false);
}

__device__ __forceinline__ v16bf combine16(v8bf lo, v8bf hi) {
  v16bf r;
#pragma unroll
  for (int i = 0; i < 8; ++i) { r[i] = lo[i]; r[i + 8] = hi[i]; }
  return r;
}

// ---------------------------------------------------------------------------
// Async global -> LDS copy (CDNA5, tracked by ASYNCcnt).
// Low 32 bits of a generic pointer into the LDS aperture == LDS byte offset
// (ISA 10.2).
// ---------------------------------------------------------------------------
#define USE_ASYNC_LDS 1

#if USE_ASYNC_LDS
__device__ __forceinline__ void async_copy_b128(const void* g, void* l) {
  unsigned int loff = (unsigned int)(uintptr_t)l;
  asm volatile("global_load_async_to_lds_b128 %0, %1, off"
               :: "v"(loff), "v"(g) : "memory");
}
__device__ __forceinline__ void wait_async0() {
  asm volatile("s_wait_asynccnt 0x0" ::: "memory");
}
#else
__device__ __forceinline__ void async_copy_b128(const void* g, void* l) {
  *(v8bf*)l = *(const v8bf*)g;
}
__device__ __forceinline__ void wait_async0() {}
#endif

// ---------------------------------------------------------------------------
// fp32 -> bf16 conversion (n must be a multiple of 4)
// ---------------------------------------------------------------------------
__global__ __launch_bounds__(256) void cvt_f32_to_bf16(
    const float* __restrict__ src, bf16_t* __restrict__ dst, int n) {
  int i = (blockIdx.x * 256 + threadIdx.x) * 4;
  if (i < n) {
    float4 v = *(const float4*)(src + i);
    v4bf o;
    o[0] = (bf16_t)v.x; o[1] = (bf16_t)v.y;
    o[2] = (bf16_t)v.z; o[3] = (bf16_t)v.w;
    *(v4bf*)(dst + i) = o;
  }
}

// ---------------------------------------------------------------------------
// bf16 WMMA GEMM: C[M,N] = A[M,K] * B[K,N]; A,B bf16 row-major, C f32 or bf16
// Block tile 128x128x32, 256 threads = 8 waves (wave32), each wave 32x64.
// Double-buffered LDS; A staged by async DMA, B pulled through registers and
// transposed-scattered after the WMMA block (software pipeline, 1 barrier/it).
// ---------------------------------------------------------------------------
#define GBM 128
#define GBN 128
#define GBK 32
#define GPAD 8   // bf16 elements of row padding (16B)

template <bool OUT_BF16>
__global__ __launch_bounds__(256) void gemm_bf16_wmma(
    const bf16_t* __restrict__ A, const bf16_t* __restrict__ B,
    void* __restrict__ Cout, int M, int N, int K) {
  __shared__ __align__(16) bf16_t As[2][GBM][GBK + GPAD];  // [buf][row][k]
  __shared__ __align__(16) bf16_t Bs[2][GBN][GBK + GPAD];  // [buf][col][k]

  const int tid  = threadIdx.x;
  const int lane = tid & 31;
  const int wave = tid >> 5;        // 0..7
  const int wr   = wave >> 1;       // 0..3 : 32-row slab
  const int wc   = wave & 1;        // 0..1 : 64-col slab
  const int hi   = (lane >> 4) & 1;
  const int l15  = lane & 15;
  const int bRow = blockIdx.y * GBM;
  const int bCol = blockIdx.x * GBN;

  // per-thread staging coordinates (constant over the loop)
  const int arow = tid >> 2;              // 0..63 (+64 for chunk 1)
  const int akc  = (tid & 3) << 3;        // 0,8,16,24
  const int bkr  = tid >> 4;              // 0..15 (+16 for chunk 1)
  const int bnc  = (tid & 15) << 3;       // 0..120
  const bf16_t* aPtr = A + (size_t)(bRow + arow) * K + akc;
  const bf16_t* bPtr = B + (size_t)bkr * N + bCol + bnc;
  const size_t aChunk = (size_t)64 * K;
  const size_t bChunk = (size_t)16 * N;

  v8bf breg[2];
  auto stageA = [&](int buf) {
    async_copy_b128(aPtr, &As[buf][arow][akc]);
    async_copy_b128(aPtr + aChunk, &As[buf][arow + 64][akc]);
  };
  auto loadB = [&]() {
    breg[0] = *(const v8bf*)bPtr;
    breg[1] = *(const v8bf*)(bPtr + bChunk);
  };
  auto scatterB = [&](int buf) {
#pragma unroll
    for (int j = 0; j < 8; ++j) Bs[buf][bnc + j][bkr] = breg[0][j];
#pragma unroll
    for (int j = 0; j < 8; ++j) Bs[buf][bnc + j][bkr + 16] = breg[1][j];
  };

  v8f acc[2][4];
#pragma unroll
  for (int mt = 0; mt < 2; ++mt)
#pragma unroll
    for (int nt = 0; nt < 4; ++nt) acc[mt][nt] = (v8f){};

  // ---- prologue: fill buffer 0 ----
  stageA(0);
  loadB();
  scatterB(0);
  aPtr += GBK; bPtr += (size_t)GBK * N;
  wait_async0();
  __syncthreads();

  const int nIter = K / GBK;
  for (int it = 0; it < nIter; ++it) {
    const int cur = it & 1, nxt = cur ^ 1;
    const bool hasNext = (it + 1) < nIter;
    if (hasNext) { stageA(nxt); loadB(); }   // DMA + global loads in flight

    // A layout: lane m=l15 holds k = ((i&8)<<1) + 8*hi + (i&7)
    v16bf afrag[2];
    const int k0a = hi * 8;
#pragma unroll
    for (int mt = 0; mt < 2; ++mt) {
      const bf16_t* p = &As[cur][wr * 32 + mt * 16 + l15][0];
      afrag[mt] = combine16(*(const v8bf*)(p + k0a),
                            *(const v8bf*)(p + k0a + 16));
    }
    // B layout: lane n=l15 holds k = ((i&8)<<1) + 8*hi + (i&7)
    v16bf bfrag[4];
    const int kb = hi * 8;
#pragma unroll
    for (int nt = 0; nt < 4; ++nt) {
      const bf16_t* p = &Bs[cur][wc * 64 + nt * 16 + l15][kb];
      bfrag[nt] = combine16(*(const v8bf*)p, *(const v8bf*)(p + 16));
    }
#pragma unroll
    for (int mt = 0; mt < 2; ++mt)
#pragma unroll
      for (int nt = 0; nt < 4; ++nt)
        acc[mt][nt] = wmma_bf16(afrag[mt], bfrag[nt], acc[mt][nt]);

    if (hasNext) {
      scatterB(nxt);                       // registers -> LDS after compute
      aPtr += GBK; bPtr += (size_t)GBK * N;
    }
    wait_async0();
    __syncthreads();
  }

  // ---- epilogue: C/D layout row m = r + 8*hi, col n = l15 ----
#pragma unroll
  for (int mt = 0; mt < 2; ++mt)
#pragma unroll
    for (int nt = 0; nt < 4; ++nt) {
      int col  = bCol + wc * 64 + nt * 16 + l15;
      int row0 = bRow + wr * 32 + mt * 16 + hi * 8;
#pragma unroll
      for (int r = 0; r < 8; ++r) {
        if (OUT_BF16)
          ((bf16_t*)Cout)[(size_t)(row0 + r) * N + col] =
              (bf16_t)acc[mt][nt][r];
        else
          ((float*)Cout)[(size_t)(row0 + r) * N + col] = acc[mt][nt][r];
      }
    }
}

// ---------------------------------------------------------------------------
// Flash attention (no mask), bf16 WMMA, fp32 accumulate, double-buffered.
// S^T = K * Q^T so each lane owns one query column: softmax reductions are
// in-register + ONE xor-16 shuffle. P/alpha live in per-wave LDS regions, so
// wave-in-order LDS needs no extra barrier (1 barrier per key-block).
// qkv: [T, 3d] bf16 (q|k|v).  y: [T, d] bf16.  hd=64, heads=16.
// ---------------------------------------------------------------------------
#define HD   64
#define BQ   64
#define BKV  64
#define APAD 8

__global__ __launch_bounds__(128) void attn_flash_wmma(
    const bf16_t* __restrict__ qkv, bf16_t* __restrict__ y,
    int T, int dModel) {
  __shared__ __align__(16) bf16_t Kt[2][BKV][HD + APAD];   // [buf][key][feat]
  __shared__ __align__(16) bf16_t Vt[2][HD][BKV + APAD];   // [buf][feat][key]
  __shared__ __align__(16) bf16_t Pw[4][16][BKV + APAD];   // [wave][query][key]
  __shared__ float alphaLds[4][16];                        // per-query factors

  const int tid  = threadIdx.x;
  const int lane = tid & 31;
  const int wave = tid >> 5;        // 0..3
  const int hi   = (lane >> 4) & 1;
  const int l15  = lane & 15;
  const int head = blockIdx.y;
  const int q0   = blockIdx.x * BQ;
  const int ld   = 3 * dModel;
  const float sscale = 0.125f;      // 1/sqrt(64)

  // per-thread staging coordinates
  const int skey = tid >> 3;              // 0..15 (+16 per chunk)
  const int sfc  = (tid & 7) << 3;        // 0..56
  const bf16_t* kPtr =
      qkv + (size_t)skey * ld + dModel + head * HD + sfc;
  const bf16_t* vPtr =
      qkv + (size_t)skey * ld + 2 * dModel + head * HD + sfc;
  const size_t cChunk = (size_t)16 * ld;

  v8bf vreg[4];
  auto stageK = [&](int buf) {
#pragma unroll
    for (int t = 0; t < 4; ++t)
      async_copy_b128(kPtr + t * cChunk, &Kt[buf][skey + t * 16][sfc]);
  };
  auto loadV = [&]() {
#pragma unroll
    for (int t = 0; t < 4; ++t) vreg[t] = *(const v8bf*)(vPtr + t * cChunk);
  };
  auto scatterV = [&](int buf) {
#pragma unroll
    for (int t = 0; t < 4; ++t)
#pragma unroll
      for (int j = 0; j < 8; ++j) Vt[buf][sfc + j][skey + t * 16] = vreg[t][j];
  };

  // Q fragments in B-matrix layout: lane n=l15 is query, k = kt*32+16*hi+i
  v16bf qbfrag[2];
  {
    const bf16_t* qrow =
        qkv + (size_t)(q0 + wave * 16 + l15) * ld + head * HD;
#pragma unroll
    for (int kt = 0; kt < 2; ++kt) {
      int base = kt * 32 + hi * 16;
      qbfrag[kt] = combine16(*(const v8bf*)(qrow + base),
                             *(const v8bf*)(qrow + base + 8));
    }
  }

  v8f oacc[4];
#pragma unroll
  for (int nt = 0; nt < 4; ++nt) oacc[nt] = (v8f){};
  float mstate = -1e30f, lstate = 0.f;  // for query l15 (dup across halves)

  // ---- prologue: fill buffer 0 ----
  stageK(0);
  loadV();
  scatterV(0);
  kPtr += (size_t)BKV * ld; vPtr += (size_t)BKV * ld;
  wait_async0();
  __syncthreads();

  const int nIter = T / BKV;
  for (int it = 0; it < nIter; ++it) {
    const int cur = it & 1, nxt = cur ^ 1;
    const bool hasNext = (it + 1) < nIter;
    if (hasNext) { stageK(nxt); loadV(); }  // next tile in flight

    // ---- S^T = K Q^T : M = keys (4x16), N = queries (16), k = feats ------
    v8f sacc[4];
#pragma unroll
    for (int ktile = 0; ktile < 4; ++ktile) sacc[ktile] = (v8f){};
#pragma unroll
    for (int kt = 0; kt < 2; ++kt)
#pragma unroll
      for (int ktile = 0; ktile < 4; ++ktile) {
        const bf16_t* p = &Kt[cur][ktile * 16 + l15][kt * 32 + hi * 16];
        v16bf a = combine16(*(const v8bf*)p, *(const v8bf*)(p + 8));
        sacc[ktile] = wmma_bf16(a, qbfrag[kt], sacc[ktile]);
      }

    // ---- online softmax: lane owns query l15, keys ktile*16 + r + 8*hi ---
    float mx = -1e30f;
#pragma unroll
    for (int ktile = 0; ktile < 4; ++ktile)
#pragma unroll
      for (int r = 0; r < 8; ++r) mx = fmaxf(mx, sacc[ktile][r]);
    mx = fmaxf(mx, __shfl_xor(mx, 16, 32));   // combine the two halves
    mx *= sscale;
    float mn    = fmaxf(mstate, mx);
    float alpha = __expf(mstate - mn);
    mstate = mn;
    float ps = 0.f;
#pragma unroll
    for (int ktile = 0; ktile < 4; ++ktile) {
      v8bf pk;
#pragma unroll
      for (int r = 0; r < 8; ++r) {
        float pv = __expf(sacc[ktile][r] * sscale - mn);
        ps += pv;
        pk[r] = (bf16_t)pv;
      }
      // P^T -> P: lane writes 8 consecutive keys of its query row
      *(v8bf*)&Pw[wave][l15][ktile * 16 + 8 * hi] = pk;
    }
    ps += __shfl_xor(ps, 16, 32);
    lstate = lstate * alpha + ps;
    if (hi == 0) alphaLds[wave][l15] = alpha;
    // Pw/alphaLds are per-wave; LDS ops of a wave are in-order -> no barrier

    // ---- rescale O by per-query alpha (rows r + 8*hi) --------------------
    float4 a0 = *(const float4*)&alphaLds[wave][8 * hi];
    float4 a1 = *(const float4*)&alphaLds[wave][8 * hi + 4];
    float al[8] = {a0.x, a0.y, a0.z, a0.w, a1.x, a1.y, a1.z, a1.w};
#pragma unroll
    for (int nt = 0; nt < 4; ++nt)
#pragma unroll
      for (int r = 0; r < 8; ++r) oacc[nt][r] *= al[r];

    // ---- O += P V : k = keys (2x32), N = features (4x16) -----------------
#pragma unroll
    for (int kt = 0; kt < 2; ++kt) {
      const bf16_t* pp = &Pw[wave][l15][kt * 32 + hi * 8];
      v16bf pa = combine16(*(const v8bf*)pp, *(const v8bf*)(pp + 16));
#pragma unroll
      for (int nt = 0; nt < 4; ++nt) {
        const bf16_t* vp = &Vt[cur][nt * 16 + l15][kt * 32 + hi * 8];
        v16bf vb = combine16(*(const v8bf*)vp, *(const v8bf*)(vp + 16));
        oacc[nt] = wmma_bf16(pa, vb, oacc[nt]);
      }
    }

    if (hasNext) {
      scatterV(nxt);                       // registers -> LDS after compute
      kPtr += (size_t)BKV * ld; vPtr += (size_t)BKV * ld;
    }
    wait_async0();
    __syncthreads();
  }

  // ---- normalize (broadcast 1/l through per-wave LDS) and store y --------
  if (hi == 0) alphaLds[wave][l15] = 1.f / lstate;
  float4 i0 = *(const float4*)&alphaLds[wave][8 * hi];
  float4 i1 = *(const float4*)&alphaLds[wave][8 * hi + 4];
  float il[8] = {i0.x, i0.y, i0.z, i0.w, i1.x, i1.y, i1.z, i1.w};
#pragma unroll
  for (int r = 0; r < 8; ++r) {
    int row = q0 + wave * 16 + r + 8 * hi;
#pragma unroll
    for (int nt = 0; nt < 4; ++nt)
      y[(size_t)row * dModel + head * HD + nt * 16 + l15] =
          (bf16_t)(oacc[nt][r] * il[r]);
  }
}

// ---------------------------------------------------------------------------
// Host-side orchestration
// ---------------------------------------------------------------------------
static inline size_t align256(size_t x) { return (x + 255) & ~size_t(255); }

extern "C" void kernel_launch(void* const* d_in, const int* in_sizes, int n_in,
                              void* d_out, int out_size, void* d_ws,
                              size_t ws_size, hipStream_t stream) {
  (void)in_sizes; (void)n_in; (void)out_size; (void)ws_size;
  const int T = 2048, D = 1024;

  const float* x     = (const float*)d_in[0];
  const float* Wqkv  = (const float*)d_in[1];
  const float* Wproj = (const float*)d_in[2];
  float* out = (float*)d_out;

  char* ws = (char*)d_ws;
  bf16_t* xb     = (bf16_t*)ws; ws += align256((size_t)T * D * 2);
  bf16_t* wqkvb  = (bf16_t*)ws; ws += align256((size_t)D * 3 * D * 2);
  bf16_t* wprojb = (bf16_t*)ws; ws += align256((size_t)D * D * 2);
  bf16_t* qkvb   = (bf16_t*)ws; ws += align256((size_t)T * 3 * D * 2);
  bf16_t* yb     = (bf16_t*)ws; ws += align256((size_t)T * D * 2);

  // 1) fp32 -> bf16
  {
    int n;
    n = T * D;
    cvt_f32_to_bf16<<<n / 4 / 256, 256, 0, stream>>>(x, xb, n);
    n = D * 3 * D;
    cvt_f32_to_bf16<<<n / 4 / 256, 256, 0, stream>>>(Wqkv, wqkvb, n);
    n = D * D;
    cvt_f32_to_bf16<<<n / 4 / 256, 256, 0, stream>>>(Wproj, wprojb, n);
  }

  // 2) qkv = x @ W_qkv   [2048,1024]x[1024,3072] -> bf16
  {
    dim3 grid(3 * D / GBN, T / GBM);
    gemm_bf16_wmma<true><<<grid, 256, 0, stream>>>(xb, wqkvb, (void*)qkvb,
                                                   T, 3 * D, D);
  }

  // 3) flash attention -> yb [2048,1024] bf16
  {
    dim3 grid(T / BQ, /*heads=*/16);
    attn_flash_wmma<<<grid, 128, 0, stream>>>(qkvb, yb, T, D);
  }

  // 4) out = y @ W_proj  [2048,1024]x[1024,1024] -> fp32
  {
    dim3 grid(D / GBN, T / GBM);
    gemm_bf16_wmma<false><<<grid, 256, 0, stream>>>(yb, wprojb, (void*)out,
                                                    T, D, D);
  }
}
